// LocalAttention_9131100471587
// MI455X (gfx1250) — matrix operand order repaired and checked by reading, not compiled, not run on hardware
//
#include <hip/hip_runtime.h>
#include <hip/hip_bf16.h>
#include <cstdint>
#include <cstddef>

#define SEQ   4096
#define DM    1024
#define HEADS 16
#define DK    64
#define HALF_WIN 128

typedef unsigned short u16;
typedef unsigned int   u32;

typedef __attribute__((ext_vector_type(16))) __bf16 v16bf;
typedef __attribute__((ext_vector_type(8)))  float  v8f;
typedef __attribute__((ext_vector_type(4)))  int    v4i;

#if defined(__HIP_DEVICE_COMPILE__) && \
    __has_builtin(__builtin_amdgcn_global_load_async_to_lds_b128) && \
    __has_builtin(__builtin_amdgcn_s_wait_asynccnt)
#define HAVE_ASYNC_LDS 1
#define AS1 __attribute__((address_space(1)))
#define AS3 __attribute__((address_space(3)))
#else
#define HAVE_ASYNC_LDS 0
#endif

// ---------------------------------------------------------------------------
// helpers
// ---------------------------------------------------------------------------
__device__ __forceinline__ u16 f2bf(float f) {
    u32 u = __builtin_bit_cast(u32, f);
    u32 r = u + 0x7FFFu + ((u >> 16) & 1u);   // round-to-nearest-even
    return (u16)(r >> 16);
}

__device__ __forceinline__ u32 pk2(u16 lo, u16 hi) {
    return (u32)lo | ((u32)hi << 16);
}

struct BPack { uint4 a, b; };   // 32 bytes == v16bf

// load 16 bf16 per lane as two 16B chunks (A/B operand fragments)
__device__ __forceinline__ v16bf ldAB(const u16* p0, const u16* p1) {
    BPack s;
    s.a = *reinterpret_cast<const uint4*>(p0);
    s.b = *reinterpret_cast<const uint4*>(p1);
    return __builtin_bit_cast(v16bf, s);
}

__device__ __forceinline__ v8f vzero8() {
    v8f v;
#pragma unroll
    for (int i = 0; i < 8; ++i) v[i] = 0.0f;
    return v;
}

__device__ __forceinline__ v8f wmma_bf16(v16bf a, v16bf b, v8f c) {
    // D = A(16x32 bf16) * B(32x16 bf16) + C(16x16 f32)
    return __builtin_amdgcn_wmma_f32_16x16x32_bf16(
        false, a, false, b, (short)0, c, false, false);
}

// ---------------------------------------------------------------------------
// elementwise f32 -> bf16
// ---------------------------------------------------------------------------
__global__ void tobf16_kernel(const float* __restrict__ src,
                              u16* __restrict__ dst, int n) {
    int stride = gridDim.x * blockDim.x;
    for (int i = blockIdx.x * blockDim.x + threadIdx.x; i < n; i += stride)
        dst[i] = f2bf(src[i]);
}

// ---------------------------------------------------------------------------
// 1024x1024 f32 weight -> transposed bf16  (WT[n][k] = W[k][n])
// ---------------------------------------------------------------------------
__global__ __launch_bounds__(256)
void wtrans_kernel(const float* __restrict__ W, u16* __restrict__ WT) {
    __shared__ float t[32][33];
    int tx = threadIdx.x & 31, ty = threadIdx.x >> 5;   // 32x8
    int x = blockIdx.x * 32 + tx;
    int y = blockIdx.y * 32 + ty;
#pragma unroll
    for (int i = 0; i < 4; ++i)
        t[ty + 8 * i][tx] = W[(size_t)(y + 8 * i) * DM + x];
    __syncthreads();
    int xo = blockIdx.y * 32 + tx;
    int yo = blockIdx.x * 32 + ty;
#pragma unroll
    for (int i = 0; i < 4; ++i)
        WT[(size_t)(yo + 8 * i) * DM + xo] = f2bf(t[tx][ty + 8 * i]);
}

// ---------------------------------------------------------------------------
// GEMM: out = A(MxK bf16, row-major) * B(KxN) + bias, B given as BT[N][K] bf16.
// Block = 4 waves sharing one 32-column strip: the 32xK B panel is staged
// through LDS once per block (async global->LDS DMA when available, double
// buffered so the DMA overlaps WMMA compute). Each wave computes a 32x32 tile:
// 4 WMMAs per K=32 step.
// MODE 0: f32 out, row-major.  MODE 1: bf16 out (scaled), row-major.
// MODE 2: bf16 out (scaled), transposed [N][M] (for V^T).
// ---------------------------------------------------------------------------
template <int MODE>
__global__ __launch_bounds__(128)
void gemm_bf16(const u16* __restrict__ A, const u16* __restrict__ BT,
               const float* __restrict__ bias, void* __restrict__ out,
               int M, int N, int K, float scale) {
    __shared__ __align__(16) u16 bbuf[2][32 * 32];   // double-buffered 32xK panel
    int lane = threadIdx.x & 31, wave = threadIdx.x >> 5;
    int lo = lane & 15, hi = lane >> 4;
    int n0 = blockIdx.x * 32;
    int m0 = (blockIdx.y * 4 + wave) * 32;

    const u16* ar0 = A + (size_t)(m0 + lo)      * K;
    const u16* ar1 = A + (size_t)(m0 + 16 + lo) * K;
    int asel = hi * 8;    // A: lanes<16 take K 0..7/16..23, lanes>=16 take 8..15/24..31
    int bsel = hi * 8;    // B: lanes<16 take K 0..7/16..23, lanes>=16 take 8..15/24..31

    // cooperative B-panel staging: 128 threads x 16B = 2KB = 32 rows x 32 halfs
    int brow = threadIdx.x >> 2;             // 0..31
    int bchk = (threadIdx.x & 3) * 8;        // half offset 0/8/16/24
    const u16* gB = BT + (size_t)(n0 + brow) * K + bchk;
    u16* lB[2] = { &bbuf[0][brow * 32 + bchk], &bbuf[1][brow * 32 + bchk] };

    auto stage = [&](int buf, int k) {
#if HAVE_ASYNC_LDS
        __builtin_amdgcn_global_load_async_to_lds_b128(
            (AS1 v4i*)(gB + k), (AS3 v4i*)lB[buf], 0, 0);
#else
        uint4 v = *reinterpret_cast<const uint4*>(gB + k);
        *reinterpret_cast<uint4*>(lB[buf]) = v;
#endif
    };

    v8f c00 = vzero8(), c01 = vzero8(), c10 = vzero8(), c11 = vzero8();

    stage(0, 0);
#if HAVE_ASYNC_LDS
    __builtin_amdgcn_s_wait_asynccnt(0);
#endif
    __syncthreads();

    for (int k = 0, i = 0; k < K; k += 32, ++i) {
        if (k + 32 < K) stage((i + 1) & 1, k + 32);   // DMA next panel during compute

        const u16* lb  = bbuf[i & 1];
        const u16* b0p = lb + (size_t)lo * 32 + bsel;
        const u16* b1p = lb + (size_t)(16 + lo) * 32 + bsel;

        v16bf a0 = ldAB(ar0 + k + asel, ar0 + k + asel + 16);
        v16bf a1 = ldAB(ar1 + k + asel, ar1 + k + asel + 16);
        v16bf b0 = ldAB(b0p, b0p + 16);
        v16bf b1 = ldAB(b1p, b1p + 16);
        c00 = wmma_bf16(a0, b0, c00);
        c01 = wmma_bf16(a0, b1, c01);
        c10 = wmma_bf16(a1, b0, c10);
        c11 = wmma_bf16(a1, b1, c11);

#if HAVE_ASYNC_LDS
        __builtin_amdgcn_s_wait_asynccnt(0);
#endif
        __syncthreads();
    }

    float bn0 = bias[n0 + lo];
    float bn1 = bias[n0 + 16 + lo];

    auto emit = [&](v8f c, int tm, int tn, float bn) {
        if (MODE == 0) {
            float* O = (float*)out;
#pragma unroll
            for (int r = 0; r < 8; ++r) {
                int gm = m0 + tm * 16 + r + 8 * hi;
                O[(size_t)gm * N + n0 + tn * 16 + lo] = c[r] + bn;
            }
        } else if (MODE == 1) {
            u16* O = (u16*)out;
#pragma unroll
            for (int r = 0; r < 8; ++r) {
                int gm = m0 + tm * 16 + r + 8 * hi;
                O[(size_t)gm * N + n0 + tn * 16 + lo] = f2bf((c[r] + bn) * scale);
            }
        } else {
            // transposed store: lane owns column gn; rows r are contiguous -> b128
            u16* O = (u16*)out;
            int gn = n0 + tn * 16 + lo;
            size_t base = (size_t)gn * M + m0 + tm * 16 + 8 * hi;
            uint4 p;
            p.x = pk2(f2bf((c[0] + bn) * scale), f2bf((c[1] + bn) * scale));
            p.y = pk2(f2bf((c[2] + bn) * scale), f2bf((c[3] + bn) * scale));
            p.z = pk2(f2bf((c[4] + bn) * scale), f2bf((c[5] + bn) * scale));
            p.w = pk2(f2bf((c[6] + bn) * scale), f2bf((c[7] + bn) * scale));
            *reinterpret_cast<uint4*>(O + base) = p;
        }
    };
    emit(c00, 0, 0, bn0);
    emit(c01, 0, 1, bn1);
    emit(c10, 1, 0, bn0);
    emit(c11, 1, 1, bn1);
}

// ---------------------------------------------------------------------------
// Flash-style local attention. One wave per (head, 16-query tile).
// Q,K row-major [SEQ][DM] bf16 (Q pre-scaled by 1/sqrt(dk)); V transposed
// [DM][SEQ] bf16. Output O row-major [SEQ][DM] bf16.
// Key tiles qt-8 .. qt+9 processed as 9 pairs (K=32 per AV WMMA).
// ---------------------------------------------------------------------------
__global__ __launch_bounds__(128)
void attn_kernel(const u16* __restrict__ Q, const u16* __restrict__ Kt,
                 const u16* __restrict__ Vt, u16* __restrict__ O) {
    __shared__ __align__(16) u16 plds[4][16 * 48];  // per-wave P staging (padded rows)
    int lane = threadIdx.x & 31, wave = threadIdx.x >> 5;
    int lo = lane & 15, hi = lane >> 4;
    int qt = blockIdx.x * 4 + wave;
    int h  = blockIdx.y;
    int q0 = qt * 16;

    const u16* qrow = Q + (size_t)(q0 + lo) * DM + h * DK;
    v16bf qa0 = ldAB(qrow + hi * 16,      qrow + hi * 16 + 8);        // dims 0..31
    v16bf qa1 = ldAB(qrow + 32 + hi * 16, qrow + 32 + hi * 16 + 8);   // dims 32..63

    float mrun[8], lrun[8];
    v8f acc[4];
#pragma unroll
    for (int d = 0; d < 4; ++d) acc[d] = vzero8();
#pragma unroll
    for (int r = 0; r < 8; ++r) { mrun[r] = -1e30f; lrun[r] = 0.0f; }

    u16* myp = plds[wave];

    for (int p = 0; p < 9; ++p) {
        int jt0 = qt - 8 + 2 * p;
        int j0a = jt0 * 16, j0b = j0a + 16;
        int j0ac = j0a < 0 ? 0 : (j0a > SEQ - 16 ? SEQ - 16 : j0a);
        int j0bc = j0b < 0 ? 0 : (j0b > SEQ - 16 ? SEQ - 16 : j0b);

        // scores: S = Q(16x64) * K_tile^T(64x16), two K=32 WMMAs per tile
        const u16* kra = Kt + (size_t)(j0ac + lo) * DM + h * DK;
        const u16* krb = Kt + (size_t)(j0bc + lo) * DM + h * DK;
        v8f s0 = vzero8(), s1 = vzero8();
        {
            v16bf kb0 = ldAB(kra + hi * 16,      kra + hi * 16 + 8);
            v16bf kb1 = ldAB(kra + 32 + hi * 16, kra + 32 + hi * 16 + 8);
            s0 = wmma_bf16(qa0, kb0, s0);
            s0 = wmma_bf16(qa1, kb1, s0);
        }
        {
            v16bf kb0 = ldAB(krb + hi * 16,      krb + hi * 16 + 8);
            v16bf kb1 = ldAB(krb + 32 + hi * 16, krb + 32 + hi * 16 + 8);
            s1 = wmma_bf16(qa0, kb0, s1);
            s1 = wmma_bf16(qa1, kb1, s1);
        }

        // mask + online softmax update (rows: VGPR index; cols: lane group of 16)
        float p0v[8], p1v[8];
#pragma unroll
        for (int r = 0; r < 8; ++r) {
            int i  = q0 + r + 8 * hi;
            int ja = j0a + lo, jb = j0b + lo;
            float sa = s0[r], sb = s1[r];
            int da = i - ja; da = da < 0 ? -da : da;
            int db = i - jb; db = db < 0 ? -db : db;
            if ((unsigned)ja >= (unsigned)SEQ || da > HALF_WIN) sa = -1e30f;
            if ((unsigned)jb >= (unsigned)SEQ || db > HALF_WIN) sb = -1e30f;

            float rm = fmaxf(sa, sb);
            rm = fmaxf(rm, __shfl_xor(rm, 1));
            rm = fmaxf(rm, __shfl_xor(rm, 2));
            rm = fmaxf(rm, __shfl_xor(rm, 4));
            rm = fmaxf(rm, __shfl_xor(rm, 8));

            float mnew = fmaxf(mrun[r], rm);
            float corr = __expf(mrun[r] - mnew);
            float pa = (sa < -1e29f) ? 0.0f : __expf(sa - mnew);
            float pb = (sb < -1e29f) ? 0.0f : __expf(sb - mnew);
            float rs = pa + pb;
            rs += __shfl_xor(rs, 1);
            rs += __shfl_xor(rs, 2);
            rs += __shfl_xor(rs, 4);
            rs += __shfl_xor(rs, 8);

            lrun[r] = lrun[r] * corr + rs;
            mrun[r] = mnew;
            acc[0][r] *= corr; acc[1][r] *= corr;
            acc[2][r] *= corr; acc[3][r] *= corr;
            p0v[r] = pa; p1v[r] = pb;
        }

        // P: C-layout -> A-layout via per-wave LDS (same-wave DS ops are in order)
#pragma unroll
        for (int r = 0; r < 8; ++r) {
            int row = r + 8 * hi;
            myp[row * 48 + lo]      = f2bf(p0v[r]);
            myp[row * 48 + lo + 16] = f2bf(p1v[r]);
        }
        asm volatile("s_wait_dscnt 0x0" ::: "memory");
        const u16* prow = myp + (size_t)lo * 48 + hi * 16;
        v16bf pA = ldAB(prow, prow + 8);   // A fragment: 16 queries x 32 keys

        // O += P * V_pair ; V^T rows are dims, keys contiguous
        int jbase = hi ? j0bc : j0ac;       // lanes<16: K 0..15 (tile0), >=16: tile1
#pragma unroll
        for (int d = 0; d < 4; ++d) {
            const u16* vr = Vt + (size_t)(h * DK + d * 16 + lo) * SEQ + jbase;
            v16bf vb = ldAB(vr, vr + 8);
            acc[d] = wmma_bf16(pA, vb, acc[d]);
        }
    }

    // normalize + store bf16 O[SEQ][DM]
#pragma unroll
    for (int r = 0; r < 8; ++r) {
        float inv = 1.0f / lrun[r];
        int gm = q0 + r + 8 * hi;
#pragma unroll
        for (int d = 0; d < 4; ++d)
            O[(size_t)gm * DM + h * DK + d * 16 + lo] = f2bf(acc[d][r] * inv);
    }
}

// ---------------------------------------------------------------------------
// launch
// ---------------------------------------------------------------------------
extern "C" void kernel_launch(void* const* d_in, const int* in_sizes, int n_in,
                              void* d_out, int out_size, void* d_ws, size_t ws_size,
                              hipStream_t stream) {
    (void)in_sizes; (void)n_in; (void)out_size; (void)ws_size;
    const float* x  = (const float*)d_in[0];
    const float* Wq = (const float*)d_in[1];
    const float* bq = (const float*)d_in[2];
    const float* Wk = (const float*)d_in[3];
    const float* bk = (const float*)d_in[4];
    const float* Wv = (const float*)d_in[5];
    const float* bv = (const float*)d_in[6];
    const float* Wo = (const float*)d_in[7];
    const float* bo = (const float*)d_in[8];

    char* ws = (char*)d_ws;
    const size_t SZX = (size_t)SEQ * DM * sizeof(u16);  // 8 MB
    const size_t SZW = (size_t)DM * DM * sizeof(u16);   // 2 MB
    u16* Xb  = (u16*)(ws);
    u16* WqT = (u16*)(ws + SZX);
    u16* WkT = (u16*)(ws + SZX + SZW);
    u16* WvT = (u16*)(ws + SZX + 2 * SZW);
    u16* WoT = (u16*)(ws + SZX + 3 * SZW);
    u16* Qb  = (u16*)(ws + SZX + 4 * SZW);
    u16* Kb  = (u16*)(ws + 2 * SZX + 4 * SZW);
    u16* Vt  = (u16*)(ws + 3 * SZX + 4 * SZW);
    u16* Ob  = (u16*)(ws + 4 * SZX + 4 * SZW);          // total 48 MB

    tobf16_kernel<<<4096, 256, 0, stream>>>(x, Xb, SEQ * DM);
    dim3 tgrid(DM / 32, DM / 32);
    wtrans_kernel<<<tgrid, 256, 0, stream>>>(Wq, WqT);
    wtrans_kernel<<<tgrid, 256, 0, stream>>>(Wk, WkT);
    wtrans_kernel<<<tgrid, 256, 0, stream>>>(Wv, WvT);
    wtrans_kernel<<<tgrid, 256, 0, stream>>>(Wo, WoT);

    dim3 ggrid(DM / 32, SEQ / 128);  // (32, 32): 4 waves/block, 32x32 tile/wave
    const float qscale = 0.125f;     // 1/sqrt(64)
    gemm_bf16<1><<<ggrid, 128, 0, stream>>>(Xb, WqT, bq, Qb, SEQ, DM, DM, qscale);
    gemm_bf16<1><<<ggrid, 128, 0, stream>>>(Xb, WkT, bk, Kb, SEQ, DM, DM, 1.0f);
    gemm_bf16<2><<<ggrid, 128, 0, stream>>>(Xb, WvT, bv, Vt, SEQ, DM, DM, 1.0f);

    attn_kernel<<<dim3(SEQ / 64, HEADS), 128, 0, stream>>>(Qb, Kb, Vt, Ob);

    gemm_bf16<0><<<ggrid, 128, 0, stream>>>(Ob, WoT, bo, (float*)d_out,
                                            SEQ, DM, DM, 1.0f);
}
